// MultiHeadAttention_128849018914
// MI455X (gfx1250) — matrix operand order repaired and checked by reading, not compiled, not run on hardware
//
#include <hip/hip_runtime.h>
#include <hip/hip_bf16.h>

typedef __attribute__((ext_vector_type(16))) _Float16 v16h;
typedef __attribute__((ext_vector_type(8)))  _Float16 v8h;
typedef __attribute__((ext_vector_type(8)))  float    v8f;
typedef __attribute__((ext_vector_type(4)))  unsigned int u32x4;
typedef __attribute__((ext_vector_type(8)))  int      i32x8;
typedef __attribute__((ext_vector_type(4)))  int      i32x4;

// Problem constants (fixed by the reference)
constexpr int kD  = 1024;     // D_MODEL
constexpr int kH  = 16;       // NUM_HEADS
constexpr int kDk = 64;       // D_K
constexpr int kB  = 2;        // batch
constexpr int kS  = 2048;     // seq len
constexpr int kM  = kB * kS;  // 4096 rows

// GEMM tiling
constexpr int BM = 64;
constexpr int BN = 128;
constexpr int BK = 32;
constexpr int LDT = 48;  // LDS row stride (elems) -> 96B, keeps 16B alignment

__device__ inline v16h load_frag16(const _Float16* p0, const _Float16* p1) {
    union { v16h v; v8h h[2]; } u;
    u.h[0] = *reinterpret_cast<const v8h*>(p0);
    u.h[1] = *reinterpret_cast<const v8h*>(p1);
    return u.v;
}

// --- CDNA5 async copy: global -> LDS, 16B per lane, tracked by ASYNCcnt ---
__device__ inline void async_cp16(_Float16* lds_dst, const _Float16* gsrc) {
    unsigned loff = (unsigned)(unsigned long long)(void*)lds_dst;  // addr[31:0] = LDS offset
    asm volatile("global_load_async_to_lds_b128 %0, %1, off"
                 :: "v"(loff), "v"(gsrc) : "memory");
}
__device__ inline void wait_async0() {
#if __has_builtin(__builtin_amdgcn_s_wait_asynccnt)
    __builtin_amdgcn_s_wait_asynccnt(0);
#else
    asm volatile("s_wait_asynccnt 0x0" ::: "memory");
#endif
}
__device__ inline void wait_tensor0() {
#if __has_builtin(__builtin_amdgcn_s_wait_tensorcnt)
    __builtin_amdgcn_s_wait_tensorcnt(0);
#else
    asm volatile("s_wait_tensorcnt 0x0" ::: "memory");
#endif
}

// --- CDNA5 TDM: load a 64(row of 64 halves) x 64(rows) f16 tile into LDS with
//     row padding 64 halves + 8 halves (stride 72) via D# pad fields. ---
__device__ inline void tdm_load_k_tile(unsigned lds_byte_off, const _Float16* gbase) {
#if __has_builtin(__builtin_amdgcn_tensor_load_to_lds)
    unsigned long long ga = (unsigned long long)gbase;
    u32x4 g0;
    g0.x = 1u;                                    // count=1 (valid user descriptor)
    g0.y = lds_byte_off;                          // lds_addr (bytes)
    g0.z = (unsigned)(ga & 0xffffffffu);          // global_addr[31:0]
    g0.w = (unsigned)((ga >> 32) & 0x01ffffffu)   // global_addr[56:32]
           | (2u << 30);                          // type = 2 ("image")
    i32x8 g1;
    // data_size=1 (2B) | pad_enable | pad_interval=4 (32 DW = 128B rows)
    // | pad_amount=3 (4 DW = 16B pad -> LDS stride 144B = 72 halves)
    g1[0] = (1 << 16) | (1 << 20) | (4 << 22) | (3 << 25);
    g1[1] = (64 << 16);            // tensor_dim0 = 64 (bits 63:48 -> dword1[31:16])
    g1[2] = 0;                     // tensor_dim0 hi16 = 0, tensor_dim1 lo16 = 0
    g1[3] = 16 | (64 << 16);       // tensor_dim1 = 1<<20 (hi16=16), tile_dim0 = 64
    g1[4] = 64;                    // tile_dim1 = 64, tile_dim2 = 0
    g1[5] = 64;                    // tensor_dim0_stride = 64 elems
    g1[6] = 0;                     // stride0 hi, tensor_dim1_stride lo (unused, 2D)
    g1[7] = 0;
    i32x4 g2 = {0, 0, 0, 0};
    i32x4 g3 = {0, 0, 0, 0};
#if defined(__clang_major__) && __clang_major__ >= 23
    i32x8 g4 = {0, 0, 0, 0, 0, 0, 0, 0};
    __builtin_amdgcn_tensor_load_to_lds(g0, g1, g2, g3, g4, 0);
#else
    __builtin_amdgcn_tensor_load_to_lds(g0, g1, g2, g3, 0);
#endif
#else
    (void)lds_byte_off; (void)gbase;
#endif
}

// ---------------------------------------------------------------------------
// Stage 1: Y = X @ W^T + b  (fp32 in), output f16 in head layout [B,H,S,d_k]
// ---------------------------------------------------------------------------
__global__ __launch_bounds__(256)
void proj_qkv_kernel(const float* __restrict__ X, const float* __restrict__ W,
                     const float* __restrict__ bias, _Float16* __restrict__ Yh) {
    __shared__ alignas(16) _Float16 As[BM][LDT];
    __shared__ alignas(16) _Float16 Bs[BN][LDT];

    const int tid  = threadIdx.x;
    const int lane = tid & 31;
    const int wave = tid >> 5;
    const int wm   = wave & 3;
    const int wn   = wave >> 2;
    const int l15  = lane & 15;
    const int hh   = lane >> 4;
    const int m0   = blockIdx.x * BM;
    const int n0   = blockIdx.y * BN;

    v8f acc[4] = {};

    for (int k0 = 0; k0 < kD; k0 += BK) {
        __syncthreads();
#pragma unroll
        for (int i = 0; i < 2; ++i) {
            int idx = tid + i * 256;
            int row = idx >> 3, cg = idx & 7;
            float4 f = reinterpret_cast<const float4*>(X + (size_t)(m0 + row) * kD + k0)[cg];
            _Float16* dst = &As[row][cg * 4];
            dst[0] = (_Float16)f.x; dst[1] = (_Float16)f.y;
            dst[2] = (_Float16)f.z; dst[3] = (_Float16)f.w;
        }
#pragma unroll
        for (int i = 0; i < 4; ++i) {
            int idx = tid + i * 256;
            int row = idx >> 3, cg = idx & 7;
            float4 f = reinterpret_cast<const float4*>(W + (size_t)(n0 + row) * kD + k0)[cg];
            _Float16* dst = &Bs[row][cg * 4];
            dst[0] = (_Float16)f.x; dst[1] = (_Float16)f.y;
            dst[2] = (_Float16)f.z; dst[3] = (_Float16)f.w;
        }
        if (k0 + BK < kD)
            __builtin_prefetch(W + (size_t)(n0 + (tid >> 3)) * kD + k0 + BK, 0, 1);
        __syncthreads();

        v16h a = load_frag16(&As[wm * 16 + l15][hh * 8],
                             &As[wm * 16 + l15][16 + hh * 8]);
#pragma unroll
        for (int t = 0; t < 4; ++t) {
            v16h b = load_frag16(&Bs[wn * 64 + t * 16 + l15][hh * 8],
                                 &Bs[wn * 64 + t * 16 + l15][16 + hh * 8]);
            acc[t] = __builtin_amdgcn_wmma_f32_16x16x32_f16(
                false, a, false, b, (short)0, acc[t], false, false);
        }
    }

#pragma unroll
    for (int t = 0; t < 4; ++t) {
        int n_g  = n0 + wn * 64 + t * 16 + l15;
        float bv = bias[n_g];
        int head = n_g >> 6;
        int cin  = n_g & 63;
#pragma unroll
        for (int r = 0; r < 8; ++r) {
            int m_g = m0 + wm * 16 + r + hh * 8;
            int bb  = m_g >> 11;
            int ss  = m_g & (kS - 1);
            size_t oidx = (((size_t)bb * kH + head) * kS + ss) * kDk + cin;
            Yh[oidx] = (_Float16)(acc[t][r] + bv);
        }
    }
}

// ---------------------------------------------------------------------------
// Stage 2: flash attention. 128 threads = 4 waves, each owns 16 query rows.
// Q preloaded via async-to-LDS copies; K tiles DMA'd by the TDM per key block.
// ---------------------------------------------------------------------------
__global__ __launch_bounds__(128)
void attn_kernel(const _Float16* __restrict__ Qh, const _Float16* __restrict__ Kh,
                 const _Float16* __restrict__ Vh, const int* __restrict__ mask,
                 _Float16* __restrict__ Oh) {
    __shared__ alignas(16) _Float16 Qs[64][72];
    __shared__ alignas(16) _Float16 Ks[64][72];
    __shared__ alignas(16) _Float16 Vt[64][72];      // transposed: [d][key]
    __shared__ alignas(16) _Float16 Pls[4][16][72];  // per-wave P tile

    const int tid  = threadIdx.x;
    const int lane = tid & 31;
    const int wave = tid >> 5;
    const int l15  = lane & 15;
    const int hh   = lane >> 4;
    const int q0   = blockIdx.x * 64;
    const int h    = blockIdx.y;
    const int b    = blockIdx.z;

    const _Float16* qptr  = Qh + (((size_t)b * kH + h) * kS + q0) * kDk;
    const _Float16* kbase = Kh + (((size_t)b * kH + h) * kS) * kDk;
    const _Float16* vbase = Vh + (((size_t)b * kH + h) * kS) * kDk;

    // Q tile (64x64 f16) via async global->LDS copies (ASYNCcnt path)
#pragma unroll
    for (int i = 0; i < 4; ++i) {
        int idx = tid + i * 128;
        int row = idx >> 3, g = idx & 7;
        async_cp16(&Qs[row][g * 8], qptr + (size_t)row * kDk + g * 8);
    }
    wait_async0();

    float mrow[8], lrow[8];
#pragma unroll
    for (int r = 0; r < 8; ++r) { mrow[r] = -1e30f; lrow[r] = 0.0f; }
    v8f oacc[4] = {};

    for (int j0 = 0; j0 < kS; j0 += 64) {
        __syncthreads();
#if __has_builtin(__builtin_amdgcn_tensor_load_to_lds)
        // K tile DMA'd by wave 0 via the Tensor Data Mover (TENSORcnt path)
        if (wave == 0) {
            tdm_load_k_tile((unsigned)(unsigned long long)(void*)&Ks[0][0],
                            kbase + (size_t)j0 * kDk);
        }
#else
        // fallback: per-lane async copies
#pragma unroll
        for (int i = 0; i < 4; ++i) {
            int idx = tid + i * 128;
            int row = idx >> 3, g = idx & 7;
            async_cp16(&Ks[row][g * 8], kbase + (size_t)(j0 + row) * kDk + g * 8);
        }
        wait_async0();
#endif
        // V tile through registers (needs transpose into Vt[d][key])
#pragma unroll
        for (int i = 0; i < 4; ++i) {
            int idx = tid + i * 128;
            int key = idx >> 3, g = idx & 7;
            union { float4 f; _Float16 hx[8]; } u;
            u.f = reinterpret_cast<const float4*>(vbase + (size_t)(j0 + key) * kDk)[g];
#pragma unroll
            for (int d = 0; d < 8; ++d) Vt[g * 8 + d][key] = u.hx[d];
        }
#if __has_builtin(__builtin_amdgcn_tensor_load_to_lds)
        if (wave == 0) wait_tensor0();
#endif
        __syncthreads();

        // scores: S = Q(16x64) * K^T, 4 key-subtiles x 2 d-chunks
        float sc[4][8];
#pragma unroll
        for (int t = 0; t < 4; ++t) {
            v8f s = {};
#pragma unroll
            for (int kc = 0; kc < 2; ++kc) {
                v16h a = load_frag16(&Qs[wave * 16 + l15][kc * 32 + hh * 8],
                                     &Qs[wave * 16 + l15][kc * 32 + 16 + hh * 8]);
                v16h bk = load_frag16(&Ks[t * 16 + l15][kc * 32 + hh * 8],
                                      &Ks[t * 16 + l15][kc * 32 + 16 + hh * 8]);
                s = __builtin_amdgcn_wmma_f32_16x16x32_f16(
                    false, a, false, bk, (short)0, s, false, false);
            }
            int kg = j0 + t * 16 + l15;
#pragma unroll
            for (int r = 0; r < 8; ++r) {
                int qg  = q0 + wave * 16 + r + hh * 8;
                float v = s[r] * 0.125f;  // 1/sqrt(64)
                if (mask[((size_t)b * kS + qg) * kS + kg] == 0) v = -1e10f;
                sc[t][r] = v;
            }
        }

        // online softmax per row (reduce over the 16 lanes of each half-wave)
        float fac[8];
#pragma unroll
        for (int r = 0; r < 8; ++r) {
            float mx = fmaxf(fmaxf(sc[0][r], sc[1][r]), fmaxf(sc[2][r], sc[3][r]));
#pragma unroll
            for (int off = 1; off < 16; off <<= 1)
                mx = fmaxf(mx, __shfl_xor(mx, off, 32));
            float mnew = fmaxf(mrow[r], mx);
            float f = __expf(mrow[r] - mnew);
            float rs = 0.0f;
#pragma unroll
            for (int t = 0; t < 4; ++t) {
                float p = __expf(sc[t][r] - mnew);
                sc[t][r] = p;
                rs += p;
            }
#pragma unroll
            for (int off = 1; off < 16; off <<= 1)
                rs += __shfl_xor(rs, off, 32);
            lrow[r] = lrow[r] * f + rs;
            mrow[r] = mnew;
            fac[r] = f;
        }
#pragma unroll
        for (int t = 0; t < 4; ++t)
#pragma unroll
            for (int r = 0; r < 8; ++r) oacc[t][r] *= fac[r];

        // round-trip P through per-wave LDS to re-fragment as a WMMA A-matrix
#pragma unroll
        for (int t = 0; t < 4; ++t)
#pragma unroll
            for (int r = 0; r < 8; ++r)
                Pls[wave][r + hh * 8][t * 16 + l15] = (_Float16)sc[t][r];

        // O += P(16x64) * V(64 x d_k)
#pragma unroll
        for (int kc = 0; kc < 2; ++kc) {
            v16h a = load_frag16(&Pls[wave][l15][kc * 32 + hh * 8],
                                 &Pls[wave][l15][kc * 32 + 16 + hh * 8]);
#pragma unroll
            for (int dt = 0; dt < 4; ++dt) {
                v16h bv = load_frag16(&Vt[dt * 16 + l15][kc * 32 + hh * 8],
                                      &Vt[dt * 16 + l15][kc * 32 + 16 + hh * 8]);
                oacc[dt] = __builtin_amdgcn_wmma_f32_16x16x32_f16(
                    false, a, false, bv, (short)0, oacc[dt], false, false);
            }
        }
    }

    // normalize, store f16 in [B,S, h*d_k + d] (heads re-concatenated)
#pragma unroll
    for (int dt = 0; dt < 4; ++dt)
#pragma unroll
        for (int r = 0; r < 8; ++r) {
            int qg = q0 + wave * 16 + r + hh * 8;
            float inv = 1.0f / lrow[r];
            Oh[((size_t)(b * kS + qg)) * kD + h * kDk + dt * 16 + l15] =
                (_Float16)(oacc[dt][r] * inv);
        }
}

// ---------------------------------------------------------------------------
// Stage 3: out = attn @ Wo^T + bo, f16 A-side, fp32 output
// ---------------------------------------------------------------------------
__global__ __launch_bounds__(256)
void proj_out_kernel(const _Float16* __restrict__ Ain, const float* __restrict__ W,
                     const float* __restrict__ bias, float* __restrict__ Out) {
    __shared__ alignas(16) _Float16 As[BM][LDT];
    __shared__ alignas(16) _Float16 Bs[BN][LDT];

    const int tid  = threadIdx.x;
    const int lane = tid & 31;
    const int wave = tid >> 5;
    const int wm   = wave & 3;
    const int wn   = wave >> 2;
    const int l15  = lane & 15;
    const int hh   = lane >> 4;
    const int m0   = blockIdx.x * BM;
    const int n0   = blockIdx.y * BN;

    v8f acc[4] = {};

    for (int k0 = 0; k0 < kD; k0 += BK) {
        __syncthreads();
        {
            int row = tid >> 2, g = tid & 3;
            float4 f = reinterpret_cast<const float4*>(Ain + (size_t)(m0 + row) * kD + k0)[g];
            *reinterpret_cast<float4*>(&As[row][g * 8]) = f;
        }
#pragma unroll
        for (int i = 0; i < 4; ++i) {
            int idx = tid + i * 256;
            int row = idx >> 3, cg = idx & 7;
            float4 f = reinterpret_cast<const float4*>(W + (size_t)(n0 + row) * kD + k0)[cg];
            _Float16* dst = &Bs[row][cg * 4];
            dst[0] = (_Float16)f.x; dst[1] = (_Float16)f.y;
            dst[2] = (_Float16)f.z; dst[3] = (_Float16)f.w;
        }
        if (k0 + BK < kD)
            __builtin_prefetch(W + (size_t)(n0 + (tid >> 3)) * kD + k0 + BK, 0, 1);
        __syncthreads();

        v16h a = load_frag16(&As[wm * 16 + l15][hh * 8],
                             &As[wm * 16 + l15][16 + hh * 8]);
#pragma unroll
        for (int t = 0; t < 4; ++t) {
            v16h b = load_frag16(&Bs[wn * 64 + t * 16 + l15][hh * 8],
                                 &Bs[wn * 64 + t * 16 + l15][16 + hh * 8]);
            acc[t] = __builtin_amdgcn_wmma_f32_16x16x32_f16(
                false, a, false, b, (short)0, acc[t], false, false);
        }
    }

#pragma unroll
    for (int t = 0; t < 4; ++t) {
        int n_g  = n0 + wn * 64 + t * 16 + l15;
        float bv = bias[n_g];
#pragma unroll
        for (int r = 0; r < 8; ++r) {
            int m_g = m0 + wm * 16 + r + hh * 8;
            Out[(size_t)m_g * kD + n_g] = acc[t][r] + bv;
        }
    }
}

// ---------------------------------------------------------------------------
extern "C" void kernel_launch(void* const* d_in, const int* in_sizes, int n_in,
                              void* d_out, int out_size, void* d_ws, size_t ws_size,
                              hipStream_t stream) {
    const float* q    = (const float*)d_in[0];
    const float* k    = (const float*)d_in[1];
    const float* v    = (const float*)d_in[2];
    const int*   mask = (const int*)d_in[3];
    const float* Wq   = (const float*)d_in[4];
    const float* bq   = (const float*)d_in[5];
    const float* Wk   = (const float*)d_in[6];
    const float* bk   = (const float*)d_in[7];
    const float* Wv   = (const float*)d_in[8];
    const float* bv   = (const float*)d_in[9];
    const float* Wo   = (const float*)d_in[10];
    const float* bo   = (const float*)d_in[11];

    const size_t elems = (size_t)kB * kS * kD;  // 4,194,304
    _Float16* ws = (_Float16*)d_ws;
    _Float16* qh = ws;
    _Float16* kh = ws + elems;
    _Float16* vh = ws + 2 * elems;
    _Float16* ao = ws + 3 * elems;

    dim3 gp(kM / BM, kD / BN);  // (64, 8)
    proj_qkv_kernel<<<gp, 256, 0, stream>>>(q, Wq, bq, qh);
    proj_qkv_kernel<<<gp, 256, 0, stream>>>(k, Wk, bk, kh);
    proj_qkv_kernel<<<gp, 256, 0, stream>>>(v, Wv, bv, vh);

    dim3 ga(kS / 64, kH, kB);   // (32, 16, 2)
    attn_kernel<<<ga, 128, 0, stream>>>(qh, kh, vh, mask, ao);

    proj_out_kernel<<<gp, 256, 0, stream>>>(ao, Wo, bo, (float*)d_out);
}
